// BiAttention_45037027066233
// MI455X (gfx1250) — matrix operand order repaired and checked by reading, not compiled, not run on hardware
//
#include <hip/hip_runtime.h>
#include <hip/hip_bf16.h>

#define Bb 16
#define Nn 1024
#define Mm 1024
#define Dd 512
#define NEGV (-2e20f)

typedef __bf16 bf16;
typedef __attribute__((ext_vector_type(16))) __bf16 v16bf;
typedef __attribute__((ext_vector_type(8)))  __bf16 v8bf;
typedef __attribute__((ext_vector_type(8)))  float  v8f;

// ---------- bf16 helpers (round-to-nearest-even) ----------
__device__ inline bf16 f2bf(float f) {
  unsigned u = __builtin_bit_cast(unsigned, f);
  u += 0x7FFFu + ((u >> 16) & 1u);
  unsigned short h = (unsigned short)(u >> 16);
  return __builtin_bit_cast(bf16, h);
}

// ---------- WMMA A-fragment loader (from global) ----------
// A fragment: 16x32 bf16 tile, row-major source with leading dim `ld`.
// Layout: lanes 0-15 -> rows 0-15, K=0..15 (elems 0-15)
//         lanes 16-31 -> rows 0-15, K=16..31.
__device__ inline v16bf load_a_frag(const bf16* tile, int ld, int lane) {
  int row = lane & 15;
  int kb  = (lane >> 4) << 4;          // 0 or 16
  const bf16* p = tile + (size_t)row * ld + kb;
  v8bf lo = *(const v8bf*)p;           // K = kb .. kb+7
  v8bf hi = *(const v8bf*)(p + 8);     // K = kb+8 .. kb+15
  v16bf r;
#pragma unroll
  for (int i = 0; i < 8; ++i) { r[i] = lo[i]; r[i + 8] = hi[i]; }
  return r;
}

// ---------- block GEMM with async-to-LDS double-buffered B tile ----------
// Block = 128 threads = 4 waves. Block output tile: 128 rows x 64 cols.
// Wave w computes rows [32w, 32w+32) x all 64 cols -> 2x4 accumulators;
// each B fragment fetched once from LDS feeds TWO WMMAs (both row groups).
// B tile in LDS: 64 rows (output cols) x 32 K bf16, row-major (64 B per row).
// Each thread DMAs 32 B of the tile with 2x global_load_async_to_lds_b128.
__device__ inline void block_gemm(const bf16* A, int lda,   // wave's 32-row A tile
                                  const bf16* Bt, int ldb,  // block's 64-row B^T tile
                                  int K, v8f acc[2][4], bf16* sm /* 4096 bf16 = 8 KB */) {
  const int t    = threadIdx.x;       // 0..127
  const int lane = t & 31;
  const unsigned smbase = (unsigned)(size_t)sm;      // LDS byte offset
  const int nT = K / 32;

  // prologue: DMA tile 0 into buffer 0
  {
    const bf16* g = Bt + (size_t)(t >> 1) * ldb + (t & 1) * 16;
    unsigned lo = smbase + (unsigned)t * 32u;
    unsigned long long ga = (unsigned long long)(size_t)g;
    asm volatile("global_load_async_to_lds_b128 %0, %1, off\n\t"
                 "global_load_async_to_lds_b128 %0, %1, off offset:16"
                 :: "v"(lo), "v"(ga) : "memory");
  }

  for (int i = 0; i < nT; ++i) {
    asm volatile("s_wait_asynccnt 0x0" ::: "memory");
    __syncthreads();   // tile i resident; everyone done reading buffer (i+1)&1

    if (i + 1 < nT) {  // DMA tile i+1 into the other buffer (overlaps with compute)
      const bf16* g = Bt + (size_t)(t >> 1) * ldb + (i + 1) * 32 + (t & 1) * 16;
      unsigned lo = smbase + (unsigned)(((i + 1) & 1) * 4096u) + (unsigned)t * 32u;
      unsigned long long ga = (unsigned long long)(size_t)g;
      asm volatile("global_load_async_to_lds_b128 %0, %1, off\n\t"
                   "global_load_async_to_lds_b128 %0, %1, off offset:16"
                   :: "v"(lo), "v"(ga) : "memory");
    }

    v16bf a0 = load_a_frag(A + i * 32, lda, lane);
    v16bf a1 = load_a_frag(A + (size_t)16 * lda + i * 32, lda, lane);
    const bf16* buf = sm + (size_t)(i & 1) * 2048;
#pragma unroll
    for (int j = 0; j < 4; ++j) {
      // B fragment from LDS: lane%16 = column N, lanes 0-15 K=0..15, 16-31 K=16..31
      int col = (lane & 15) + j * 16;
      int kb  = (lane >> 4) << 4;
      v16bf b = *(const v16bf*)(buf + (size_t)col * 32 + kb);
      acc[0][j] = __builtin_amdgcn_wmma_f32_16x16x32_bf16(
          false, a0, false, b, (short)0, acc[0][j], false, false);
      acc[1][j] = __builtin_amdgcn_wmma_f32_16x16x32_bf16(
          false, a1, false, b, (short)0, acc[1][j], false, false);
    }
  }
}

// ---------- prep kernels ----------
__global__ __launch_bounds__(256) void prep_x1_kernel(
    const float* __restrict__ x1, const float* __restrict__ w,
    bf16* __restrict__ Abf, bf16* __restrict__ x1T, float* __restrict__ s1) {
  int wave = threadIdx.x >> 5, lane = threadIdx.x & 31;
  int rowFlat = blockIdx.x * 8 + wave;      // 0 .. B*N-1
  int b = rowFlat >> 10, n = rowFlat & 1023;
  const float* xr = x1 + (size_t)rowFlat * Dd;
  const float* w1 = w;
  const float* w3 = w + 2 * Dd;
  float dot = 0.f;
  for (int i = lane; i < Dd; i += 32) {
    float v = xr[i];
    dot += v * w1[i];
    Abf[(size_t)rowFlat * Dd + i] = f2bf(v * w3[i]);
    x1T[((size_t)b * Dd + i) * Nn + n] = f2bf(v);
  }
#pragma unroll
  for (int off = 16; off > 0; off >>= 1) dot += __shfl_xor(dot, off, 32);
  if (lane == 0) s1[rowFlat] = dot;
}

__global__ __launch_bounds__(256) void prep_x2_kernel(
    const float* __restrict__ x2, const float* __restrict__ w,
    bf16* __restrict__ x2bf, bf16* __restrict__ x2T, float* __restrict__ s2) {
  int wave = threadIdx.x >> 5, lane = threadIdx.x & 31;
  int rowFlat = blockIdx.x * 8 + wave;      // 0 .. B*M-1
  int b = rowFlat >> 10, m = rowFlat & 1023;
  const float* xr = x2 + (size_t)rowFlat * Dd;
  const float* w2 = w + Dd;
  float dot = 0.f;
  for (int i = lane; i < Dd; i += 32) {
    float v = xr[i];
    dot += v * w2[i];
    x2bf[(size_t)rowFlat * Dd + i] = f2bf(v);
    x2T[((size_t)b * Dd + i) * Mm + m] = f2bf(v);
  }
#pragma unroll
  for (int off = 16; off > 0; off >>= 1) dot += __shfl_xor(dot, off, 32);
  if (lane == 0) s2[rowFlat] = dot;
}

// ---------- sim GEMM: sim = Abf @ x2^T + s1 + s2 + bias, x2-mask -> NEG ----------
__global__ __launch_bounds__(128) void sim_gemm_kernel(
    const bf16* __restrict__ Abf,     // (B, N, D)
    const bf16* __restrict__ X2bf,    // (B, M, D)  == B^T rows
    const float* __restrict__ s1,     // (B, N)
    const float* __restrict__ s2,     // (B, M)
    const int* __restrict__ x2_mask,  // (B, M)
    const float* __restrict__ biasp,
    float* __restrict__ sim) {        // (B, N, M)
  __shared__ __align__(64) bf16 smem[4096];
  int b    = blockIdx.z;
  int row0 = blockIdx.y * 128 + (threadIdx.x >> 5) * 32;
  int col0 = blockIdx.x * 64;
  int lane = threadIdx.x & 31;
  const bf16* A  = Abf  + ((size_t)b * Nn + row0) * Dd;
  const bf16* Bt = X2bf + ((size_t)b * Mm + col0) * Dd;
  v8f acc[2][4] = {};
  block_gemm(A, Dd, Bt, Dd, Dd, acc, smem);

  float bias = biasp[0];
  int rsub = (lane >> 4) * 8;          // C/D layout: M = r + 8*(lane>=16)
#pragma unroll
  for (int rg = 0; rg < 2; ++rg) {
    int rowg = row0 + rg * 16;
    float s1v[8];
#pragma unroll
    for (int r = 0; r < 8; ++r) s1v[r] = s1[b * Nn + rowg + rsub + r];
#pragma unroll
    for (int j = 0; j < 4; ++j) {
      int col = col0 + j * 16 + (lane & 15);
      float s2v = s2[b * Mm + col];
      bool msk = x2_mask[b * Mm + col] != 0;
      float* out = sim + ((size_t)b * Nn + rowg + rsub) * Mm + col;
#pragma unroll
      for (int r = 0; r < 8; ++r) {
        float v = acc[rg][j][r] + s1v[r] + s2v + bias;
        out[(size_t)r * Mm] = msk ? NEGV : v;
      }
    }
  }
}

// ---------- softmax kernels ----------
__global__ __launch_bounds__(256) void row_softmax_kernel(
    const float* __restrict__ sim, bf16* __restrict__ Prow) {
  __shared__ float red[256];
  int row = blockIdx.x;                       // b*N + n
  int t = threadIdx.x;
  const float* p = sim + (size_t)row * Mm;
  float v[4];
  float mx = -3.4e38f;
#pragma unroll
  for (int i = 0; i < 4; ++i) { v[i] = p[t + i * 256]; mx = fmaxf(mx, v[i]); }
  red[t] = mx; __syncthreads();
  for (int s = 128; s > 0; s >>= 1) { if (t < s) red[t] = fmaxf(red[t], red[t + s]); __syncthreads(); }
  mx = red[0]; __syncthreads();
  float e[4], sum = 0.f;
#pragma unroll
  for (int i = 0; i < 4; ++i) { e[i] = __expf(v[i] - mx); sum += e[i]; }
  red[t] = sum; __syncthreads();
  for (int s = 128; s > 0; s >>= 1) { if (t < s) red[t] += red[t + s]; __syncthreads(); }
  float inv = 1.0f / red[0];
  bf16* o = Prow + (size_t)row * Mm;
#pragma unroll
  for (int i = 0; i < 4; ++i) o[t + i * 256] = f2bf(e[i] * inv);
}

__global__ __launch_bounds__(256) void col_softmax_kernel(
    const float* __restrict__ sim, const int* __restrict__ x1_mask,
    bf16* __restrict__ PcolT) {
  __shared__ float red[256];
  int bm = blockIdx.x;                        // b*M + m
  int b = bm >> 10, m = bm & 1023;
  int t = threadIdx.x;
  const float* p = sim + ((size_t)b * Nn) * Mm + m;   // stride Mm down the column
  const int* msk = x1_mask + b * Nn;
  float v[4];
  float mx = -3.4e38f;
#pragma unroll
  for (int i = 0; i < 4; ++i) {
    int n = t + i * 256;
    float x = p[(size_t)n * Mm];
    v[i] = (msk[n] != 0) ? NEGV : x;
    mx = fmaxf(mx, v[i]);
  }
  red[t] = mx; __syncthreads();
  for (int s = 128; s > 0; s >>= 1) { if (t < s) red[t] = fmaxf(red[t], red[t + s]); __syncthreads(); }
  mx = red[0]; __syncthreads();
  float e[4], sum = 0.f;
#pragma unroll
  for (int i = 0; i < 4; ++i) { e[i] = __expf(v[i] - mx); sum += e[i]; }
  red[t] = sum; __syncthreads();
  for (int s = 128; s > 0; s >>= 1) { if (t < s) red[t] += red[t + s]; __syncthreads(); }
  float inv = 1.0f / red[0];
  bf16* o = PcolT + ((size_t)b * Mm + m) * Nn;        // transposed store: contiguous in n
#pragma unroll
  for (int i = 0; i < 4; ++i) o[t + i * 256] = f2bf(e[i] * inv);
}

// ---------- generic GEMMs ----------
// Out f32 (B, rowsTot, colsTot) = A (B, rowsTot, K) @ Bt (B, colsTot, K)^T
__global__ __launch_bounds__(128) void gemm_f32out_kernel(
    const bf16* __restrict__ Aall, const bf16* __restrict__ Btall,
    float* __restrict__ Out, int K, int rowsTot, int colsTot) {
  __shared__ __align__(64) bf16 smem[4096];
  int b    = blockIdx.z;
  int row0 = blockIdx.y * 128 + (threadIdx.x >> 5) * 32;
  int col0 = blockIdx.x * 64;
  int lane = threadIdx.x & 31;
  const bf16* A  = Aall  + ((size_t)b * rowsTot + row0) * K;
  const bf16* Bt = Btall + ((size_t)b * colsTot + col0) * K;
  v8f acc[2][4] = {};
  block_gemm(A, K, Bt, K, K, acc, smem);
  int rsub = (lane >> 4) * 8;
#pragma unroll
  for (int rg = 0; rg < 2; ++rg) {
    int rowg = row0 + rg * 16;
#pragma unroll
    for (int j = 0; j < 4; ++j) {
      int col = col0 + j * 16 + (lane & 15);
      float* out = Out + ((size_t)b * rowsTot + rowg + rsub) * colsTot + col;
#pragma unroll
      for (int r = 0; r < 8; ++r) out[(size_t)r * colsTot] = acc[rg][j][r];
    }
  }
}

// OutT bf16 (B, colsTot, rowsTot) = transpose of (A @ Bt^T), stores contiguous per lane.
__global__ __launch_bounds__(128) void gemm_bf16T_kernel(
    const bf16* __restrict__ Aall, const bf16* __restrict__ Btall,
    bf16* __restrict__ OutT, int K, int rowsTot, int colsTot) {
  __shared__ __align__(64) bf16 smem[4096];
  int b    = blockIdx.z;
  int row0 = blockIdx.y * 128 + (threadIdx.x >> 5) * 32;
  int col0 = blockIdx.x * 64;
  int lane = threadIdx.x & 31;
  const bf16* A  = Aall  + ((size_t)b * rowsTot + row0) * K;
  const bf16* Bt = Btall + ((size_t)b * colsTot + col0) * K;
  v8f acc[2][4] = {};
  block_gemm(A, K, Bt, K, K, acc, smem);
  int rsub = (lane >> 4) * 8;
#pragma unroll
  for (int rg = 0; rg < 2; ++rg) {
    int rowg = row0 + rg * 16;
#pragma unroll
    for (int j = 0; j < 4; ++j) {
      int col = col0 + j * 16 + (lane & 15);
      bf16* out = OutT + ((size_t)b * colsTot + col) * rowsTot + rowg + rsub;
#pragma unroll
      for (int r = 0; r < 8; ++r) out[r] = f2bf(acc[rg][j][r]);
    }
  }
}

// ---------- launcher ----------
extern "C" void kernel_launch(void* const* d_in, const int* in_sizes, int n_in,
                              void* d_out, int out_size, void* d_ws, size_t ws_size,
                              hipStream_t stream) {
  (void)in_sizes; (void)n_in; (void)out_size; (void)ws_size;
  const float* x1      = (const float*)d_in[0];
  const int*   x1_mask = (const int*)d_in[1];
  const float* x2      = (const float*)d_in[2];
  const int*   x2_mask = (const int*)d_in[3];
  const float* w       = (const float*)d_in[4];
  const float* bias    = (const float*)d_in[5];
  float* attn_a = (float*)d_out;
  float* attn_b = attn_a + (size_t)Bb * Nn * Dd;

  char* ws = (char*)d_ws;
  size_t off = 0;
  auto alloc = [&](size_t bytes) {
    char* p = ws + off;
    off += (bytes + 255) & ~(size_t)255;
    return (void*)p;
  };
  bf16*  Abf   = (bf16*) alloc((size_t)Bb * Nn * Dd * 2);  // bf16(x1*w3)
  bf16*  x1T   = (bf16*) alloc((size_t)Bb * Nn * Dd * 2);  // (B, D, N)
  bf16*  x2bf  = (bf16*) alloc((size_t)Bb * Mm * Dd * 2);  // (B, M, D)
  bf16*  x2T   = (bf16*) alloc((size_t)Bb * Mm * Dd * 2);  // (B, D, M)
  float* s1    = (float*)alloc((size_t)Bb * Nn * 4);
  float* s2    = (float*)alloc((size_t)Bb * Mm * 4);
  float* sim   = (float*)alloc((size_t)Bb * Nn * Mm * 4);  // (B, N, M)
  bf16*  Prow  = (bf16*) alloc((size_t)Bb * Nn * Mm * 2);  // (B, N, M)
  bf16*  PcolT = (bf16*) alloc((size_t)Bb * Nn * Mm * 2);  // (B, M, N)
  bf16*  q2cT  = (bf16*) alloc((size_t)Bb * Dd * Mm * 2);  // (B, D, M)

  prep_x1_kernel<<<Bb * Nn / 8, 256, 0, stream>>>(x1, w, Abf, x1T, s1);
  prep_x2_kernel<<<Bb * Mm / 8, 256, 0, stream>>>(x2, w, x2bf, x2T, s2);

  sim_gemm_kernel<<<dim3(Mm / 64, Nn / 128, Bb), 128, 0, stream>>>(
      Abf, x2bf, s1, s2, x2_mask, bias, sim);

  row_softmax_kernel<<<Bb * Nn, 256, 0, stream>>>(sim, Prow);
  col_softmax_kernel<<<Bb * Mm, 256, 0, stream>>>(sim, x1_mask, PcolT);

  // attn_a = Prow @ x2      : (N x M) x (M x D)
  gemm_f32out_kernel<<<dim3(Dd / 64, Nn / 128, Bb), 128, 0, stream>>>(
      Prow, x2T, attn_a, Mm, Nn, Dd);
  // q2cT = (PcolT @ x1)^T   : (M x N) x (N x D) stored (D, M)
  gemm_bf16T_kernel<<<dim3(Dd / 64, Mm / 128, Bb), 128, 0, stream>>>(
      PcolT, x1T, q2cT, Nn, Mm, Dd);
  // attn_b = Prow @ q2c     : (N x M) x (M x D)
  gemm_f32out_kernel<<<dim3(Dd / 64, Nn / 128, Bb), 128, 0, stream>>>(
      Prow, q2cT, attn_b, Mm, Nn, Dd);
}
